// GRU_72395968741891
// MI455X (gfx1250) — matrix operand order repaired and checked by reading, not compiled, not run on hardware
//
#include <hip/hip_runtime.h>
#include <hip/hip_bf16.h>
#include <cstdint>
#include <cstddef>

// ---------------------------------------------------------------------------
// Problem constants (match reference)
// ---------------------------------------------------------------------------
#define BB    256          // batch
#define TT    250          // timesteps
#define NSTK  2048         // stocks (layer-0 input dim / head output dim)
#define HGRU  256          // hidden
#define GGRU  768          // 3*H gate dim
#define NSEL  50
#define REBAL 32
#define UBW   0.1f
#define LBW   0.0f

typedef __attribute__((ext_vector_type(16))) __bf16 v16bf;
typedef __attribute__((ext_vector_type(8)))  __bf16 v8bf;
typedef __attribute__((ext_vector_type(8)))  float  v8f;

__device__ __forceinline__ v16bf cat8(v8bf lo, v8bf hi) {
  return __builtin_shufflevector(lo, hi, 0,1,2,3,4,5,6,7,8,9,10,11,12,13,14,15);
}

// ---------------------------------------------------------------------------
// f32 -> bf16 weight conversion (one-shot, tiny)
// ---------------------------------------------------------------------------
__global__ __launch_bounds__(256) void cvt_bf16_kernel(const float* __restrict__ s,
                                                       __bf16* __restrict__ d, int n) {
  int i = blockIdx.x * 256 + threadIdx.x;
  if (i < n) d[i] = (__bf16)s[i];
}

// ---------------------------------------------------------------------------
// Generic GEMM:  out[r*ldc + g] = epi( sum_k A[aoff(r)+k] * Wb[g*K+k] + bias[g] )
//   aoff(r) = (r % Q)*S1 + (r / Q)*S2    (handles [B,T,*] row remapping)
// WG tile 128(M) x 128(N), 8 waves = 4(M) x 2(N), wave tile 32x64.
// A converted f32->bf16 while staging to LDS (VALU path);
// W staged by CDNA5 async global->LDS DMA (ASYNCcnt) -- pure bf16 copy.
// EPI: 0 = none, 1 = sigmoid, 2 = silu
// ---------------------------------------------------------------------------
template<int EPI>
__global__ __launch_bounds__(256) void gemm_bias_epi(
    const float* __restrict__ A, const __bf16* __restrict__ Wb,
    const float* __restrict__ bias, float* __restrict__ out,
    int K, int Q, long long S1, long long S2, int ldc)
{
  __shared__ __align__(16) __bf16 As[128 * 40];   // 128 rows x 32 k (stride 40)
  __shared__ __align__(16) __bf16 Ws[128 * 40];   // 128 cols x 32 k (stride 40)

  const int tid  = threadIdx.x;
  const int lane = tid & 31;
  const int wave = tid >> 5;
  const int wm   = wave & 3;          // 0..3  (M quarter)
  const int wn   = wave >> 2;         // 0..1  (N half)
  const long long r0 = (long long)blockIdx.y * 128;
  const int g0 = blockIdx.x * 128;

  const int mrow = lane & 15;
  const int hf   = lane >> 4;

  v8f acc[2][4];
#pragma unroll
  for (int i = 0; i < 2; ++i)
#pragma unroll
    for (int j = 0; j < 4; ++j)
#pragma unroll
      for (int p = 0; p < 8; ++p) acc[i][j][p] = 0.f;

  // staging assignment: thread handles half-row (16 k) of both A and W tiles
  const int srow = tid >> 1;            // 0..127
  const int skh  = (tid & 1) * 16;      // 0 or 16
  const long long rr = r0 + srow;
  const long long aoff = (rr % Q) * S1 + (rr / Q) * S2;
  const size_t woff = (size_t)(g0 + srow) * (size_t)K;
  const unsigned ldsW = (unsigned)(uintptr_t)&Ws[srow * 40 + skh];

  for (int kb = 0; kb < K; kb += 32) {
    __syncthreads();                    // protect LDS from prev-iter fragment reads
    // ---- stage W via async global->LDS DMA (CDNA5, ASYNCcnt-tracked) ----
    // NB: the 24-bit instruction offset is added to BOTH the global address
    // and the LDS address (CDNA5 ISA), so one offset:16
    // copies the second 16B of the 32B chunk.
    {
      const __bf16* gW = Wb + woff + kb + skh;
      asm volatile(
          "global_load_async_to_lds_b128 %0, %1, off\n\t"
          "global_load_async_to_lds_b128 %0, %1, off offset:16"
          :: "v"(ldsW), "v"(gW));
    }
    // ---- stage A (f32 -> bf16, VALU convert) ----
    {
      const float* ap = A + aoff + kb + skh;
      float t0[16];
      *(float4*)&t0[0]  = *(const float4*)(ap + 0);
      *(float4*)&t0[4]  = *(const float4*)(ap + 4);
      *(float4*)&t0[8]  = *(const float4*)(ap + 8);
      *(float4*)&t0[12] = *(const float4*)(ap + 12);
      v8bf lo, hi;
#pragma unroll
      for (int e = 0; e < 8; ++e) { lo[e] = (__bf16)t0[e]; hi[e] = (__bf16)t0[8 + e]; }
      v8bf* ad = (v8bf*)&As[srow * 40 + skh];
      ad[0] = lo; ad[1] = hi;
    }
    if (kb + 32 < K) {
      __builtin_prefetch((const char*)(A + aoff + kb + 32 + skh), 0, 1);
    }
    asm volatile("s_wait_asynccnt 0x0" ::: "memory");   // our async LDS writes done
    __syncthreads();                                    // all waves' staging visible

    // ---- fragments (ISA 16-bit A/B VGPR layouts) ----
    v16bf afr[2], bfr[4];
#pragma unroll
    for (int i = 0; i < 2; ++i) {
      const __bf16* pA = &As[(wm * 32 + i * 16 + mrow) * 40 + (hf << 3)];
      afr[i] = cat8(*(const v8bf*)pA, *(const v8bf*)(pA + 16));
    }
#pragma unroll
    for (int j = 0; j < 4; ++j) {
      const __bf16* pB = &Ws[(wn * 64 + j * 16 + mrow) * 40 + (hf << 3)];
      bfr[j] = cat8(*(const v8bf*)pB, *(const v8bf*)(pB + 16));
    }
#pragma unroll
    for (int i = 0; i < 2; ++i)
#pragma unroll
      for (int j = 0; j < 4; ++j)
        acc[i][j] = __builtin_amdgcn_wmma_f32_16x16x32_bf16(
            false, afr[i], false, bfr[j], (short)0, acc[i][j], false, false);
  }

  // ---- epilogue: bias + activation, scatter per C-layout (row = hf*8+p, col = lane&15)
#pragma unroll
  for (int i = 0; i < 2; ++i) {
    const long long rb = r0 + wm * 32 + i * 16 + hf * 8;
#pragma unroll
    for (int j = 0; j < 4; ++j) {
      const int gc = g0 + wn * 64 + j * 16 + mrow;
      const float bv = bias[gc];
#pragma unroll
      for (int p = 0; p < 8; ++p) {
        float v = acc[i][j][p] + bv;
        if (EPI == 1) v = 1.f / (1.f + __expf(-v));
        else if (EPI == 2) v = v / (1.f + __expf(-v));
        out[(rb + p) * (long long)ldc + gc] = v;
      }
    }
  }
}

// ---------------------------------------------------------------------------
// GRU recurrent scan. One WG = 16 batch rows for all T steps (no cross-WG dep).
// 512 threads = 16 waves (4/SIMD); each wave owns 3 WMMA N-tiles (48 of 768
// gate cols). All Whh B-fragments (3 tiles x 8 k-steps = 192 VGPRs) are
// preloaded and stay register-resident for all 250 steps; launch_bounds(512,1)
// budgets VGPRs for that (grid = 16 WGs on separate WGPs, occupancy is
// grid-bound, not VGPR-bound). h kept f32 in LDS; A fragments built by
// on-the-fly f32->bf16 conversion.
// ---------------------------------------------------------------------------
__global__ __launch_bounds__(512, 1) void gru_scan(
    const float* __restrict__ xg,      // [T, B, 768] (bih already added)
    const __bf16* __restrict__ Whh,    // [768, 256] bf16
    const float* __restrict__ bhh,     // [768]
    float* __restrict__ out_seq,       // [B, T, H] or nullptr
    float* __restrict__ out_last)      // [B, H]    or nullptr
{
  __shared__ __align__(16) float hCur[16 * 260];   // [16][H] padded
  __shared__ __align__(16) float hg[16 * 776];     // [16][768] padded

  const int tid  = threadIdx.x;
  const int lane = tid & 31;
  const int wave = tid >> 5;          // 0..15
  const int b0   = blockIdx.x * 16;
  const int mrow = lane & 15;
  const int hf   = lane >> 4;

  // ---- preload all recurrent-weight fragments for this wave's 3 tiles ----
  v16bf bw[3][8];
#pragma unroll
  for (int jj = 0; jj < 3; ++jj) {
    const int g = (wave * 3 + jj) * 16 + mrow;
#pragma unroll
    for (int kk = 0; kk < 8; ++kk) {
      const v8bf* wp = (const v8bf*)(Whh + (size_t)g * HGRU + kk * 32 + (hf << 3));
      bw[jj][kk] = cat8(wp[0], wp[2]);
    }
  }

  for (int i = tid; i < 16 * 260; i += 512) hCur[i] = 0.f;
  __syncthreads();

  for (int t = 0; t < TT; ++t) {
    v8f acc[3];
#pragma unroll
    for (int jj = 0; jj < 3; ++jj)
#pragma unroll
      for (int p = 0; p < 8; ++p) acc[jj][p] = 0.f;

#pragma unroll
    for (int kk = 0; kk < 8; ++kk) {
      const float* hp = &hCur[mrow * 260 + kk * 32 + (hf << 3)];
      v16bf af;
#pragma unroll
      for (int e = 0; e < 8; ++e) { af[e] = (__bf16)hp[e]; af[e + 8] = (__bf16)hp[e + 16]; }
#pragma unroll
      for (int jj = 0; jj < 3; ++jj)
        acc[jj] = __builtin_amdgcn_wmma_f32_16x16x32_bf16(
            false, af, false, bw[jj][kk], (short)0, acc[jj], false, false);
    }
#pragma unroll
    for (int jj = 0; jj < 3; ++jj) {
      const int gc = (wave * 3 + jj) * 16 + mrow;
#pragma unroll
      for (int p = 0; p < 8; ++p) hg[(hf * 8 + p) * 776 + gc] = acc[jj][p];
    }
    __syncthreads();

    // fused gates + state update (4096 units / 512 threads = 8 uniform iters)
    for (int idx = tid; idx < 16 * HGRU; idx += 512) {
      const int row = idx >> 8, u = idx & 255;
      const float* xgp = xg + ((long long)t * BB + (b0 + row)) * GGRU;
      const float hr = hg[row * 776 + u]           + bhh[u];
      const float hz = hg[row * 776 + 256 + u]     + bhh[256 + u];
      const float hn = hg[row * 776 + 512 + u]     + bhh[512 + u];
      const float xr = xgp[u], xz = xgp[256 + u], xn = xgp[512 + u];
      const float r  = 1.f / (1.f + __expf(-(xr + hr)));
      const float z  = 1.f / (1.f + __expf(-(xz + hz)));
      const float nn = tanhf(xn + r * hn);
      const float h  = hCur[row * 260 + u];
      const float hnew = (1.f - z) * nn + z * h;
      hCur[row * 260 + u] = hnew;
      if (out_seq)  out_seq[((long long)(b0 + row) * TT + t) * HGRU + u] = hnew;
      if (out_last && t == TT - 1) out_last[(b0 + row) * HGRU + u] = hnew;
    }
    __syncthreads();
  }
}

// ---------------------------------------------------------------------------
// block reductions (256 threads)
// ---------------------------------------------------------------------------
__device__ __forceinline__ float blk_sum(float* red, int tid, float v) {
  __syncthreads(); red[tid] = v; __syncthreads();
#pragma unroll
  for (int s = 128; s > 0; s >>= 1) { if (tid < s) red[tid] += red[tid + s]; __syncthreads(); }
  return red[0];
}
__device__ __forceinline__ float blk_max(float* red, int tid, float v) {
  __syncthreads(); red[tid] = v; __syncthreads();
#pragma unroll
  for (int s = 128; s > 0; s >>= 1) { if (tid < s) red[tid] = fmaxf(red[tid], red[tid + s]); __syncthreads(); }
  return red[0];
}
__device__ __forceinline__ int blk_or(int* red, int tid, int v) {
  __syncthreads(); red[tid] = v; __syncthreads();
#pragma unroll
  for (int s = 128; s > 0; s >>= 1) { if (tid < s) red[tid] |= red[tid + s]; __syncthreads(); }
  return red[0];
}

// ---------------------------------------------------------------------------
// Per-row: softmax(silu logits), top-50 on sigmoid(att), mask+normalize,
// 32-iteration capped-simplex rebalance (exact reference semantics).
// ---------------------------------------------------------------------------
__global__ __launch_bounds__(256) void select_rebalance(
    const float* __restrict__ att, const float* __restrict__ logits,
    float* __restrict__ outw)
{
  __shared__ float wts[NSTK];
  __shared__ float wv[NSTK];
  __shared__ float oldv[NSTK];
  __shared__ unsigned char selm[NSTK];
  __shared__ float redf[256];
  __shared__ int   redi[256];
  __shared__ int   flag;

  const int tid = threadIdx.x;
  const int b = blockIdx.x;
  const float* arow = att + (size_t)b * NSTK;
  const float* lrow = logits + (size_t)b * NSTK;

  // softmax over (already silu'd) logits
  float lm = -1e30f;
  for (int i = tid; i < NSTK; i += 256) lm = fmaxf(lm, lrow[i]);
  const float mx = blk_max(redf, tid, lm);
  float ps = 0.f;
  for (int i = tid; i < NSTK; i += 256) { float e = __expf(lrow[i] - mx); wts[i] = e; ps += e; }
  const float sm = blk_sum(redf, tid, ps);
  const float inv = 1.f / sm;
  for (int i = tid; i < NSTK; i += 256) { wts[i] *= inv; selm[i] = 0; }
  __syncthreads();

  // top-50 on att by repeated argmax extraction
  for (int it = 0; it < NSEL; ++it) {
    float bv = -1e30f; int bi = -1;
    for (int i = tid; i < NSTK; i += 256)
      if (!selm[i]) { float v = arow[i]; if (v > bv) { bv = v; bi = i; } }
    redf[tid] = bv; redi[tid] = bi; __syncthreads();
    for (int s = 128; s > 0; s >>= 1) {
      if (tid < s) {
        float v2 = redf[tid + s]; int i2 = redi[tid + s];
        if (v2 > redf[tid] || (v2 == redf[tid] && i2 >= 0 && i2 < redi[tid])) {
          redf[tid] = v2; redi[tid] = i2;
        }
      }
      __syncthreads();
    }
    if (tid == 0 && redi[0] >= 0) selm[redi[0]] = 1;
    __syncthreads();
  }

  // mask + renormalize
  float msum = 0.f;
  for (int i = tid; i < NSTK; i += 256) { float mval = selm[i] ? wts[i] : 0.f; wv[i] = mval; msum += mval; }
  const float tot = blk_sum(redf, tid, msum);
  const float dn = 1.f / (tot + 1e-8f);
  for (int i = tid; i < NSTK; i += 256) {
    const float w0 = wv[i] * dn;
    const int sel = w0 > 0.f;
    selm[i] = (unsigned char)sel;
    const float o = sel ? w0 : 0.f;
    oldv[i] = o;
    wv[i] = fminf(fmaxf(o, LBW), UBW);
  }
  if (tid == 0) flag = 0;
  __syncthreads();

  for (int it = 0; it < REBAL; ++it) {
    const int done = flag;
    float lv = 0.f, nsv = 0.f; int anyn = 0;
    for (int i = tid; i < NSTK; i += 256) {
      const int nm = (wv[i] != UBW) && selm[i];
      anyn |= nm;
      lv  += oldv[i] - wv[i];
      nsv += nm ? wv[i] : 0.f;
    }
    const float leftover = blk_sum(redf, tid, lv);
    const float nsum = blk_sum(redf, tid, nsv);
    const int hasnom = blk_or(redi, tid, anyn);
    const int upd = (!done) && hasnom;
    const float dnm = leftover / ((nsum == 0.f) ? 1.f : nsum);
    if (upd) {
      for (int i = tid; i < NSTK; i += 256) {
        const int nm = (wv[i] != UBW) && selm[i];
        const float w1 = wv[i] + (nm ? dnm * wv[i] : 0.f);
        wv[i] = w1; oldv[i] = w1;
      }
    }
    __syncthreads();
    int ov = 0;
    for (int i = tid; i < NSTK; i += 256) ov |= (wv[i] > UBW);
    const int over = blk_or(redi, tid, ov);
    if (upd && over)
      for (int i = tid; i < NSTK; i += 256) wv[i] = fminf(fmaxf(wv[i], LBW), UBW);
    if (tid == 0) {
      int d = done;
      if (!done && !hasnom) d = 1;
      if (upd && !over) d = 1;
      flag = d;
    }
    __syncthreads();
  }
  for (int i = tid; i < NSTK; i += 256) outw[(size_t)b * NSTK + i] = wv[i];
}

// ---------------------------------------------------------------------------
// Host launcher (graph-capture safe: only kernel launches on `stream`)
// ---------------------------------------------------------------------------
extern "C" void kernel_launch(void* const* d_in, const int* in_sizes, int n_in,
                              void* d_out, int out_size, void* d_ws, size_t ws_size,
                              hipStream_t stream) {
  const float* x    = (const float*)d_in[0];
  const float* Wih0 = (const float*)d_in[1];
  const float* Whh0 = (const float*)d_in[2];
  const float* bih0 = (const float*)d_in[3];
  const float* bhh0 = (const float*)d_in[4];
  const float* Wih1 = (const float*)d_in[5];
  const float* Whh1 = (const float*)d_in[6];
  const float* bih1 = (const float*)d_in[7];
  const float* bhh1 = (const float*)d_in[8];
  const float* Wa   = (const float*)d_in[9];
  const float* ba   = (const float*)d_in[10];
  const float* Wf   = (const float*)d_in[11];
  const float* bfv  = (const float*)d_in[12];

  char* p = (char*)d_ws;
  auto carve = [&](size_t bytes) -> void* {
    void* r = (void*)p;
    p += (bytes + 255) & ~(size_t)255;
    return r;
  };
  float* xg    = (float*)carve((size_t)TT * BB * GGRU * 4);   // 196.6 MB (reused L0/L1)
  float* hseq  = (float*)carve((size_t)BB * TT * HGRU * 4);   // 65.5 MB
  float* hlast = (float*)carve((size_t)BB * HGRU * 4);
  float* attb  = (float*)carve((size_t)BB * NSTK * 4);
  float* logb  = (float*)carve((size_t)BB * NSTK * 4);
  __bf16* wih0b = (__bf16*)carve((size_t)GGRU * NSTK * 2);
  __bf16* whh0b = (__bf16*)carve((size_t)GGRU * HGRU * 2);
  __bf16* wih1b = (__bf16*)carve((size_t)GGRU * HGRU * 2);
  __bf16* whh1b = (__bf16*)carve((size_t)GGRU * HGRU * 2);
  __bf16* wab   = (__bf16*)carve((size_t)NSTK * HGRU * 2);
  __bf16* wfb   = (__bf16*)carve((size_t)NSTK * HGRU * 2);

  auto cvt = [&](const float* s, __bf16* d, int n) {
    cvt_bf16_kernel<<<(n + 255) / 256, 256, 0, stream>>>(s, d, n);
  };
  cvt(Wih0, wih0b, GGRU * NSTK);
  cvt(Whh0, whh0b, GGRU * HGRU);
  cvt(Wih1, wih1b, GGRU * HGRU);
  cvt(Whh1, whh1b, GGRU * HGRU);
  cvt(Wa,   wab,   NSTK * HGRU);
  cvt(Wf,   wfb,   NSTK * HGRU);

  // layer-0 input projection: rows r = t*B+b, A row = x[b,t,:]
  gemm_bias_epi<0><<<dim3(GGRU / 128, (TT * BB) / 128), 256, 0, stream>>>(
      x, wih0b, bih0, xg, NSTK, BB, (long long)TT * NSTK, (long long)NSTK, GGRU);
  // layer-0 scan -> hseq
  gru_scan<<<BB / 16, 512, 0, stream>>>(xg, whh0b, bhh0, hseq, nullptr);
  // layer-1 input projection: A row = hseq[b,t,:]
  gemm_bias_epi<0><<<dim3(GGRU / 128, (TT * BB) / 128), 256, 0, stream>>>(
      hseq, wih1b, bih1, xg, HGRU, BB, (long long)TT * HGRU, (long long)HGRU, GGRU);
  // layer-1 scan -> hlast only
  gru_scan<<<BB / 16, 512, 0, stream>>>(xg, whh1b, bhh1, nullptr, hlast);
  // heads: att = sigmoid(h @ Wa^T + ba), logits = silu(h @ Wf^T + bf)
  gemm_bias_epi<1><<<dim3(NSTK / 128, BB / 128), 256, 0, stream>>>(
      hlast, wab, ba, attb, HGRU, BB, (long long)HGRU, 0LL, NSTK);
  gemm_bias_epi<2><<<dim3(NSTK / 128, BB / 128), 256, 0, stream>>>(
      hlast, wfb, bfv, logb, HGRU, BB, (long long)HGRU, 0LL, NSTK);
  // top-50 + softmax mask + rebalance
  select_rebalance<<<BB, 256, 0, stream>>>(attb, logb, (float*)d_out);
}
